// ResidualRVQ_51238959841871
// MI455X (gfx1250) — matrix operand order repaired and checked by reading, not compiled, not run on hardware
//
#include <hip/hip_runtime.h>
#include <hip/hip_bf16.h>

// Problem constants (from reference): B=128, D=128, T=1024, C=8, K=1024.
#define B_ 128
#define D_ 128
#define T_ 1024
#define C_ 8
#define K_ 1024
#define NWG 64           // workgroups in cooperative main kernel (K-slice of 16 codes each)
#define NTHREADS 256     // 8 wave32 per workgroup
#define RLD 136          // padded LDS row stride (halves): 272B = 17*16B -> aligned + conflict-free

typedef __attribute__((ext_vector_type(16))) __bf16 v16bf;
typedef __attribute__((ext_vector_type(8)))  float  v8f;
typedef __attribute__((ext_vector_type(4)))  int    v4i;

union ABU { v16bf v; unsigned int u[8]; };

// ---- optional CDNA5 async global->LDS path (guarded; falls back to sync stage) ----
#if defined(__AMDGCN__) && __has_builtin(__builtin_amdgcn_global_load_async_to_lds_b128)
#define USE_ASYNC_LDS 1
typedef __attribute__((address_space(1))) v4i glob_v4i;  // AS1 (prints as __device__)
typedef __attribute__((address_space(3))) v4i lds_v4i;   // AS3 (prints as __shared__)
#endif

__device__ __forceinline__ void wait_async_le1(void) {
#ifdef USE_ASYNC_LDS
#if __has_builtin(__builtin_amdgcn_s_wait_asynccnt)
  __builtin_amdgcn_s_wait_asynccnt(1);
#else
  asm volatile("s_wait_asynccnt 0x1" ::: "memory");
#endif
#endif
}
__device__ __forceinline__ void wait_async_le0(void) {
#ifdef USE_ASYNC_LDS
#if __has_builtin(__builtin_amdgcn_s_wait_asynccnt)
  __builtin_amdgcn_s_wait_asynccnt(0);
#else
  asm volatile("s_wait_asynccnt 0x0" ::: "memory");
#endif
#endif
}

// Stage one 16-code x 128-d bf16 codebook slice (4KB payload) into LDS with a
// padded row stride. 256 threads x 16B = exactly one b128 per lane.
__device__ __forceinline__ void stage_slice(const unsigned short* gsrc,
                                            unsigned short* lbuf, int tid) {
  int code = tid >> 4, chunk = tid & 15;  // 16 codes x 16 chunks of 16B
#ifdef USE_ASYNC_LDS
  __builtin_amdgcn_global_load_async_to_lds_b128(
      (glob_v4i*)(gsrc + code * D_ + chunk * 8),
      (lds_v4i*)(lbuf + code * RLD + chunk * 8), 0, 0);
#else
  uint4 v = *(const uint4*)(gsrc + code * D_ + chunk * 8);
  *(uint4*)(lbuf + code * RLD + chunk * 8) = v;
#endif
}

__device__ __forceinline__ float bf2f(unsigned short h) {
  return __uint_as_float(((unsigned)h) << 16);
}
__device__ __forceinline__ unsigned short f2bf(float f) {
  unsigned u = __float_as_uint(f);
  return (unsigned short)((u + 0x7FFFu + ((u >> 16) & 1u)) >> 16);
}

// A-matrix fragment (16x32 bf16), per CDNA5 ISA layout:
// lanes 0-15: VGPR v<4 -> K=2v,2v+1 ; v>=4 -> K=16+2(v-4).. ; lanes 16-31 offset +8.
__device__ __forceinline__ v16bf load_a16(const unsigned short* p, int ld, int m0,
                                          int kb, int lane) {
  ABU r;
  const unsigned short* base = p + (m0 + (lane & 15)) * ld;
  int hk = (lane >> 4) ? 8 : 0;
#pragma unroll
  for (int v = 0; v < 4; ++v)
    r.u[v] = *(const unsigned int*)(base + kb + hk + 2 * v);
#pragma unroll
  for (int v = 0; v < 4; ++v)
    r.u[4 + v] = *(const unsigned int*)(base + kb + 16 + hk + 2 * v);
  return r.v;
}

// B-matrix fragment (32x16 bf16), operand stored row-major by N (each row = one
// dot-product operand vector of length D): same K order as A: VGPR v<4 -> K=2v,2v+1;
// v>=4 -> K=16+2(v-4)..; lanes 16-31 offset +8, column n = n0 + (lane&15).
__device__ __forceinline__ v16bf load_b16(const unsigned short* p, int ld, int n0,
                                          int kb, int lane) {
  ABU r;
  const unsigned short* base = p + (n0 + (lane & 15)) * ld;
  int hk = (lane >> 4) ? 8 : 0;
#pragma unroll
  for (int v = 0; v < 4; ++v)
    r.u[v] = *(const unsigned int*)(base + kb + hk + 2 * v);
#pragma unroll
  for (int v = 0; v < 4; ++v)
    r.u[4 + v] = *(const unsigned int*)(base + kb + 16 + hk + 2 * v);
  return r.v;
}

// Sense-reversal grid barrier (all NWG workgroups resident; MI455X has >> 64 WGPs).
__device__ __forceinline__ void grid_barrier(unsigned int* bar) {
  __syncthreads();
  __threadfence();
  if (threadIdx.x == 0) {
    unsigned int gen = __hip_atomic_load(&bar[1], __ATOMIC_ACQUIRE, __HIP_MEMORY_SCOPE_AGENT);
    unsigned int arrived =
        __hip_atomic_fetch_add(&bar[0], 1u, __ATOMIC_ACQ_REL, __HIP_MEMORY_SCOPE_AGENT);
    if (arrived == NWG - 1) {
      __hip_atomic_store(&bar[0], 0u, __ATOMIC_RELAXED, __HIP_MEMORY_SCOPE_AGENT);
      __hip_atomic_fetch_add(&bar[1], 1u, __ATOMIC_ACQ_REL, __HIP_MEMORY_SCOPE_AGENT);
    } else {
      while (__hip_atomic_load(&bar[1], __ATOMIC_ACQUIRE, __HIP_MEMORY_SCOPE_AGENT) == gen)
        __builtin_amdgcn_s_sleep(2);
    }
  }
  __syncthreads();
  __threadfence();
}

// ---------------- preprocessing kernels ----------------

__global__ void k_init(unsigned long long* g_win, unsigned short* g_state,
                       unsigned int* bar) {
  int tid = threadIdx.x;
  if (tid == 0) bar[0] = 0;  // gen (bar[1]) may be arbitrary: read-then-compare protocol
  for (int i = tid; i < 2 * C_ * B_; i += 256) g_win[i] = ~0ULL;
  for (int i = tid; i < 2 * B_ * D_; i += 256) g_state[i] = 0;
}

// codebooks f32 [C*K][D] -> bf16 copy + per-row squared norm
__global__ void k_prep_cb(const float* __restrict__ cb, unsigned short* __restrict__ cbB,
                          float* __restrict__ cnorm) {
  int row = blockIdx.x;   // 0..C*K-1
  int d = threadIdx.x;    // 0..127
  float v = cb[row * D_ + d];
  cbB[row * D_ + d] = f2bf(v);
  float p = v * v;
#pragma unroll
  for (int off = 16; off > 0; off >>= 1) p += __shfl_down(p, off, 32);
  __shared__ float part[4];
  if ((threadIdx.x & 31) == 0) part[threadIdx.x >> 5] = p;
  __syncthreads();
  if (threadIdx.x == 0) cnorm[row] = part[0] + part[1] + part[2] + part[3];
}

// predictor_w [D][D][3] -> bf16 W0,W1 stored row-major [j][d] (B-operand layout)
__global__ void k_prep_w(const float* __restrict__ pw, unsigned short* __restrict__ w0b,
                         unsigned short* __restrict__ w1b) {
  int i = blockIdx.x * 256 + threadIdx.x;  // i = j*128 + d < 16384
  w0b[i] = f2bf(pw[i * 3 + 0]);
  w1b[i] = f2bf(pw[i * 3 + 1]);
}

// mel [B][D][T] f32 -> melT [T][B*D] bf16 (tiled transpose through LDS)
__global__ void k_prep_mel(const float* __restrict__ mel, unsigned short* __restrict__ melT) {
  __shared__ float tile[32][33];
  int r0 = blockIdx.y * 32;  // row = b*128 + d, 0..16383
  int t0 = blockIdx.x * 32;  // t
#pragma unroll
  for (int j = 0; j < 32; j += 8)
    tile[threadIdx.y + j][threadIdx.x] =
        mel[(unsigned)(r0 + threadIdx.y + j) * T_ + t0 + threadIdx.x];
  __syncthreads();
#pragma unroll
  for (int j = 0; j < 32; j += 8) {
    float v = tile[threadIdx.x][threadIdx.y + j];
    melT[(unsigned)(t0 + threadIdx.y + j) * (B_ * D_) + r0 + threadIdx.x] = f2bf(v);
  }
}

// ---------------- cooperative main kernel ----------------

__global__ __launch_bounds__(NTHREADS) void rvq_main(
    const unsigned short* __restrict__ melT,   // [T][B*D] bf16
    const unsigned short* __restrict__ cbB,    // [C*K][D] bf16
    const float* __restrict__ cnorm,           // [C*K]
    const unsigned short* __restrict__ w0b,    // [D][D] bf16
    const unsigned short* __restrict__ w1b,    // [D][D] bf16
    const float* __restrict__ pb,              // [D]
    const float* __restrict__ cbF32,           // [C*K][D] f32 (original)
    float* __restrict__ g_pred,                // [B*D]
    unsigned short* __restrict__ g_state,      // [2][B*D] bf16 (mel_q history)
    unsigned long long* __restrict__ g_win,    // [2][C][B] packed (key<<32|code)
    unsigned int* __restrict__ bar,            // grid barrier {count, gen}
    float* __restrict__ outMelq,               // [B][D][T]
    float* __restrict__ outIdx,                // [T][C][B] as float
    float* __restrict__ outCommit, float* __restrict__ outUtil) {
  __shared__ unsigned short rLds[B_ * RLD];     // residual, bf16, padded rows (34.0 KB)
  __shared__ unsigned short cbLds[2][16 * RLD]; // double-buffered codebook slice (8.5 KB)
  __shared__ float predLds[B_ * 16];            // pred slice (wg<8)
  __shared__ float qsumLds[B_ * 16];            // q_sum slice (wg<8)
  __shared__ unsigned int kWin[B_];
  __shared__ unsigned int bitmap[K_ / 32];
  __shared__ float red[NTHREADS];

  const int wg = blockIdx.x;
  const int tid = threadIdx.x;
  const int wave = tid >> 5;
  const int lane = tid & 31;
  const int m0 = wave * 16;        // each wave owns a 16-row M tile
  const int codeBase = wg * 16;    // this WG's 16-code K-slice

  float commitAcc = 0.0f, utilAcc = 0.0f;  // meaningful on wg0/thread0 only

  for (int t = 0; t < T_; ++t) {
    const int par = t & 1;

    // Reset next-parity winner slots (safe: last read >= 1 full step ago).
    if (wg == 8) {
      for (int i = tid; i < C_ * B_; i += NTHREADS) g_win[(1 - par) * C_ * B_ + i] = ~0ULL;
    }

    // ---- Phase P: predictor GEMM, N-sliced over WGs 0..7 ----
    if (wg < 8) {
      v8f acc = {};
      if (t > 0) {
        const unsigned short* p2 = g_state + (size_t)(t & 1) * (B_ * D_);        // prev2
        const unsigned short* p1 = g_state + (size_t)((t - 1) & 1) * (B_ * D_);  // prev1
#pragma unroll
        for (int kb = 0; kb < D_; kb += 32) {
          v16bf a = load_a16(p2, D_, m0, kb, lane);
          v16bf b = load_b16(w0b, D_, wg * 16, kb, lane);
          acc = __builtin_amdgcn_wmma_f32_16x16x32_bf16(false, a, false, b, (short)0, acc,
                                                        false, false);
        }
#pragma unroll
        for (int kb = 0; kb < D_; kb += 32) {
          v16bf a = load_a16(p1, D_, m0, kb, lane);
          v16bf b = load_b16(w1b, D_, wg * 16, kb, lane);
          acc = __builtin_amdgcn_wmma_f32_16x16x32_bf16(false, a, false, b, (short)0, acc,
                                                        false, false);
        }
      }
      const int ncol = lane & 15;
      const float bias = (t > 0) ? pb[wg * 16 + ncol] : 0.0f;  // t==0: pred is exactly 0
#pragma unroll
      for (int v = 0; v < 8; ++v) {
        int mrow = m0 + ((lane >> 4) << 3) + v;
        float pv = acc[v] + bias;
        predLds[mrow * 16 + ncol] = pv;
        g_pred[mrow * D_ + wg * 16 + ncol] = pv;
      }
      for (int i = tid; i < B_ * 16; i += NTHREADS) qsumLds[i] = 0.0f;
    }
    grid_barrier(bar);

    // Kick the async stage of codebook slice 0 so it overlaps residual build.
    stage_slice(cbB + (size_t)codeBase * D_, cbLds[0], tid);

    // ---- Build residual r = mel_t - pred in LDS (bf16) ----
    {
      const unsigned short* frame = melT + (size_t)t * (B_ * D_);
      for (int i = tid; i < B_ * D_; i += NTHREADS) {
        float rv = bf2f(frame[i]) - g_pred[i];
        rLds[(i >> 7) * RLD + (i & 127)] = f2bf(rv);
      }
    }

    // ---- 8 residual-VQ stages (codebook slice double-buffered through LDS) ----
    for (int c = 0; c < C_; ++c) {
      if (c + 1 < C_)  // prefetch-copy next slice while this one is consumed
        stage_slice(cbB + (size_t)((c + 1) * K_ + codeBase) * D_, cbLds[(c + 1) & 1], tid);
      if (c + 2 < C_)  // pull the one after toward L2
        __builtin_prefetch(&cbB[((size_t)(c + 2) * K_ + codeBase) * D_], 0, 0);
      if (c + 1 < C_) wait_async_le1();  // slice c resident; slice c+1 may be in flight
      else            wait_async_le0();
      __syncthreads();

      // Distance GEMM: S = r @ cb_slice^T  (argmin key = ||c||^2 - 2 S; ||r||^2 dropped)
      v8f acc = {};
      const unsigned short* cbSlice = cbLds[c & 1];
#pragma unroll
      for (int kb = 0; kb < D_; kb += 32) {
        v16bf a = load_a16(rLds, RLD, m0, kb, lane);
        v16bf b = load_b16(cbSlice, RLD, 0, kb, lane);
        acc = __builtin_amdgcn_wmma_f32_16x16x32_bf16(false, a, false, b, (short)0, acc,
                                                      false, false);
      }
      float cn = cnorm[c * K_ + codeBase + (lane & 15)];
#pragma unroll
      for (int v = 0; v < 8; ++v) {
        float dist = cn - 2.0f * acc[v];
        unsigned ud = __float_as_uint(dist);
        ud = (ud & 0x80000000u) ? ~ud : (ud | 0x80000000u);  // monotone float->uint
        unsigned long long key =
            (((unsigned long long)ud) << 32) | (unsigned)(codeBase + (lane & 15));
#pragma unroll
        for (int off = 1; off < 16; off <<= 1) {
          unsigned long long o = __shfl_xor(key, off, 32);
          key = (o < key) ? o : key;
        }
        if ((lane & 15) == 0) {
          int row = m0 + ((lane >> 4) << 3) + v;
          atomicMin(&g_win[par * C_ * B_ + c * B_ + row], key);
        }
      }
      grid_barrier(bar);

      // Stage winners; wg0 also emits indices + utilization bitmap.
      if (tid < B_) {
        unsigned long long w = g_win[par * C_ * B_ + c * B_ + tid];
        unsigned k = (unsigned)(w & 0xFFFFFFFFu);
        kWin[tid] = k;
        if (wg == 0) outIdx[(t * C_ + c) * B_ + tid] = (float)k;
      }
      if (wg == 0 && tid < K_ / 32) bitmap[tid] = 0u;
      __syncthreads();
      if (wg == 0 && tid < B_)
        atomicOr(&bitmap[kWin[tid] >> 5], 1u << (kWin[tid] & 31));

      // r -= q (every WG keeps its own copy current); wg<8 accumulate q_sum slice.
      float mySq = 0.0f;
      for (int i = tid; i < B_ * D_; i += NTHREADS) {
        int b = i >> 7, d = i & 127;
        unsigned k = kWin[b];
        float rv = bf2f(rLds[b * RLD + d]) - bf2f(cbB[((size_t)(c * K_ + k)) * D_ + d]);
        rLds[b * RLD + d] = f2bf(rv);
        mySq += rv * rv;
      }
      if (wg < 8) {
        for (int i = tid; i < B_ * 16; i += NTHREADS) {
          int b = i >> 4, dd = i & 15;
          qsumLds[i] += cbF32[((size_t)(c * K_ + kWin[b])) * D_ + wg * 16 + dd];
        }
      }
      __syncthreads();

      // commit = mean((r_c - q_c)^2) = mean(r_next^2); util = unique-hit fraction.
      red[tid] = mySq;
      __syncthreads();
#pragma unroll
      for (int s = NTHREADS / 2; s > 0; s >>= 1) {
        if (tid < s) red[tid] += red[tid + s];
        __syncthreads();
      }
      if (wg == 0 && tid == 0) {
        commitAcc += red[0] * (1.0f / (B_ * D_));
        int cnt = 0;
        for (int wdx = 0; wdx < K_ / 32; ++wdx) cnt += __popc(bitmap[wdx]);
        utilAcc += (float)cnt * (1.0f / K_);
      }
      __syncthreads();
    }

    // ---- Emit mel_q_t = pred + q_sum; roll quantized state (WGs 0..7 cover D) ----
    if (wg < 8) {
      for (int i = tid; i < B_ * 16; i += NTHREADS) {
        int b = i >> 4, dg = wg * 16 + (i & 15);
        float mq = predLds[b * 16 + (i & 15)] + qsumLds[i];
        outMelq[((size_t)b * D_ + dg) * T_ + t] = mq;
        g_state[(size_t)par * (B_ * D_) + b * D_ + dg] = f2bf(mq);
      }
    }
    grid_barrier(bar);
  }

  if (wg == 0 && tid == 0) {
    *outCommit = commitAcc * (1.0f / (C_ * T_));
    *outUtil = utilAcc * (1.0f / (C_ * T_));
  }
}

// ---------------- host launcher ----------------

extern "C" void kernel_launch(void* const* d_in, const int* in_sizes, int n_in,
                              void* d_out, int out_size, void* d_ws, size_t ws_size,
                              hipStream_t stream) {
  (void)in_sizes; (void)n_in; (void)out_size; (void)ws_size;
  const float* mel = (const float*)d_in[0];      // [B,D,T]
  const float* pw  = (const float*)d_in[1];      // [D,D,3]
  const float* pbv = (const float*)d_in[2];      // [D]
  const float* cb  = (const float*)d_in[3];      // [C,K,D]

  char* ws = (char*)d_ws;
  size_t off = 0;
  auto take = [&](size_t bytes) {
    void* p = ws + off;
    off += (bytes + 255) & ~(size_t)255;
    return p;
  };
  unsigned short* melT   = (unsigned short*)take((size_t)T_ * B_ * D_ * 2);   // 32 MB
  unsigned short* cbB    = (unsigned short*)take((size_t)C_ * K_ * D_ * 2);   // 2 MB
  float*          cnorm  = (float*)take((size_t)C_ * K_ * 4);
  unsigned short* w0b    = (unsigned short*)take((size_t)D_ * D_ * 2);
  unsigned short* w1b    = (unsigned short*)take((size_t)D_ * D_ * 2);
  float*          g_pred = (float*)take((size_t)B_ * D_ * 4);
  unsigned short* g_state= (unsigned short*)take((size_t)2 * B_ * D_ * 2);
  unsigned long long* g_win = (unsigned long long*)take((size_t)2 * C_ * B_ * 8);
  unsigned int*   bar    = (unsigned int*)take(256);

  float* out = (float*)d_out;
  float* outMelq = out;                                   // B*D*T
  float* outIdx  = out + (size_t)B_ * D_ * T_;            // T*C*B
  float* outCom  = outIdx + (size_t)T_ * C_ * B_;
  float* outUtl  = outCom + 1;

  k_init<<<1, 256, 0, stream>>>(g_win, g_state, bar);
  k_prep_cb<<<C_ * K_, 128, 0, stream>>>(cb, cbB, cnorm);
  k_prep_w<<<(D_ * D_) / 256, 256, 0, stream>>>(pw, w0b, w1b);
  k_prep_mel<<<dim3(T_ / 32, (B_ * D_) / 32), dim3(32, 8), 0, stream>>>(mel, melT);
  rvq_main<<<NWG, NTHREADS, 0, stream>>>(melT, cbB, cnorm, w0b, w1b, pbv, cb, g_pred,
                                         g_state, g_win, bar, outMelq, outIdx, outCom,
                                         outUtl);
}
